// GRUCell_87703232184582
// MI455X (gfx1250) — matrix operand order repaired and checked by reading, not compiled, not run on hardware
//
#include <hip/hip_runtime.h>

typedef __bf16 bf16_t;
typedef __attribute__((ext_vector_type(4)))  __bf16 v4bf;
typedef __attribute__((ext_vector_type(8)))  __bf16 v8bf;
typedef __attribute__((ext_vector_type(16))) __bf16 v16bf;
typedef __attribute__((ext_vector_type(8)))  float  v8f;

#define GRU_I 512
#define GRU_H 512
#define GRU_K 1024          // I + H
#define GRU_B 16384
#define LDA2  528           // padded LDS row stride (bf16 elems) for 512-col phase

#define WMMA_BF16(A, Bm, C) \
  __builtin_amdgcn_wmma_f32_16x16x32_bf16(false, (A), false, (Bm), (short)0, (C), false, false)

__device__ __forceinline__ float fast_sigmoid(float x) {
  return __builtin_amdgcn_rcpf(1.0f + __expf(-x));
}
__device__ __forceinline__ float fast_tanh(float x) {
  return 2.0f * __builtin_amdgcn_rcpf(1.0f + __expf(-2.0f * x)) - 1.0f;
}

// ---------------- fp32 -> bf16 weight conversion (3 MB, L2-resident) ----------------
__global__ __launch_bounds__(256) void gru_cvt_weights(
    const float* __restrict__ wg, const float* __restrict__ wi,
    const float* __restrict__ wh, bf16_t* __restrict__ dst) {
  const long NG = (long)GRU_K * (2 * GRU_H);   // 1024*1024
  const long NI = (long)GRU_I * GRU_H;         // 512*512
  const long total = NG + 2 * NI;
  long i = (long)blockIdx.x * 256 + threadIdx.x;
  if (i >= total) return;
  float v;
  if (i < NG)           v = wg[i];
  else if (i < NG + NI) v = wi[i - NG];
  else                  v = wh[i - NG - NI];
  dst[i] = (bf16_t)v;
}

// ---------------- fused GRU cell: 3 GEMMs (bf16 WMMA) + gates ----------------
// Block: 256 threads (8 waves), 32 batch rows; each wave: 2 row-tiles x 2 feature-tiles.
// Each B-fragment pair feeds TWO WMMAs (M-reuse) -> halves L2 weight traffic.
__global__ __launch_bounds__(256) void gru_fused_wmma(
    const float* __restrict__ input,  const float* __restrict__ hidden,
    const bf16_t* __restrict__ Wg,    const bf16_t* __restrict__ Wi,
    const bf16_t* __restrict__ Wh,
    const float* __restrict__ b_gate, const float* __restrict__ b_i,
    const float* __restrict__ b_h,    float* __restrict__ out) {
  __shared__ bf16_t smA[32 * LDA2];          // 32 rows x 512 cols (one K-phase), ~33.8 KB

  const int  tid = threadIdx.x;
  const long m0  = (long)blockIdx.x * 32;

  const int lane = tid & 31;
  const int wave = tid >> 5;
  const int j    = lane & 15;
  const int half = lane >> 4;
  const int offA = half * 16;                // A frag contiguous K block per half-wave
  const int offB = half * 16;                // B frag contiguous K block per half-wave

  // feature tiles: nt = blockIdx.y*16 + wave + 8*t,  t in {0,1}
  const bf16_t* wgr[2]; const bf16_t* wgz[2];
  const bf16_t* wir[2]; const bf16_t* whr[2];
  int nidx[2];
#pragma unroll
  for (int t = 0; t < 2; ++t) {
    const int n = (blockIdx.y * 16 + wave + 8 * t) * 16 + j;
    nidx[t] = n;
    wgr[t] = Wg + (long)n * GRU_K;
    wgz[t] = Wg + (long)(GRU_H + n) * GRU_K;
    wir[t] = Wi + (long)n * GRU_I;
    whr[t] = Wh + (long)n * GRU_H;
  }

  v8f ar[2][2] = {}, az[2][2] = {}, ai[2][2] = {}, ah[2][2] = {};

  const bf16_t* smrow0 = smA + j * LDA2;
  const bf16_t* smrow1 = smA + (16 + j) * LDA2;

  const int srow = tid >> 7;                 // staging: 2 rows per pass
  const int scol = (tid & 127) * 4;

  // ================= phase 1: K in [0,512) -> A = input, feeds r,z,i =================
#pragma unroll
  for (int p = 0; p < 16; ++p) {
    const int row = 2 * p + srow;
    float4 f = *(const float4*)(input + (m0 + row) * GRU_I + scol);
    v4bf b = { (bf16_t)f.x, (bf16_t)f.y, (bf16_t)f.z, (bf16_t)f.w };
    *(v4bf*)(smA + row * LDA2 + scol) = b;
  }
  __syncthreads();

  for (int k = 0; k < GRU_I; k += 32) {
    v16bf a0, a1;
    ((v8bf*)&a0)[0] = *(const v8bf*)(smrow0 + k + offA);
    ((v8bf*)&a0)[1] = *(const v8bf*)(smrow0 + k + offA + 8);
    ((v8bf*)&a1)[0] = *(const v8bf*)(smrow1 + k + offA);
    ((v8bf*)&a1)[1] = *(const v8bf*)(smrow1 + k + offA + 8);
#pragma unroll
    for (int t = 0; t < 2; ++t) {
      v16bf br, bz, bx;
      ((v8bf*)&br)[0] = *(const v8bf*)(wgr[t] + k + offB);
      ((v8bf*)&br)[1] = *(const v8bf*)(wgr[t] + k + offB + 8);
      ((v8bf*)&bz)[0] = *(const v8bf*)(wgz[t] + k + offB);
      ((v8bf*)&bz)[1] = *(const v8bf*)(wgz[t] + k + offB + 8);
      ((v8bf*)&bx)[0] = *(const v8bf*)(wir[t] + k + offB);
      ((v8bf*)&bx)[1] = *(const v8bf*)(wir[t] + k + offB + 8);
      ar[0][t] = WMMA_BF16(a0, br, ar[0][t]);
      ar[1][t] = WMMA_BF16(a1, br, ar[1][t]);
      az[0][t] = WMMA_BF16(a0, bz, az[0][t]);
      az[1][t] = WMMA_BF16(a1, bz, az[1][t]);
      ai[0][t] = WMMA_BF16(a0, bx, ai[0][t]);
      ai[1][t] = WMMA_BF16(a1, bx, ai[1][t]);
    }
  }
  __syncthreads();   // everyone done reading phase-1 tile

  // ================= phase 2: K in [512,1024) -> A = hidden, feeds r,z,h =============
#pragma unroll
  for (int p = 0; p < 16; ++p) {
    const int row = 2 * p + srow;
    float4 f = *(const float4*)(hidden + (m0 + row) * GRU_H + scol);
    v4bf b = { (bf16_t)f.x, (bf16_t)f.y, (bf16_t)f.z, (bf16_t)f.w };
    *(v4bf*)(smA + row * LDA2 + scol) = b;
  }
  __syncthreads();

  for (int k = 0; k < GRU_H; k += 32) {
    v16bf a0, a1;
    ((v8bf*)&a0)[0] = *(const v8bf*)(smrow0 + k + offA);
    ((v8bf*)&a0)[1] = *(const v8bf*)(smrow0 + k + offA + 8);
    ((v8bf*)&a1)[0] = *(const v8bf*)(smrow1 + k + offA);
    ((v8bf*)&a1)[1] = *(const v8bf*)(smrow1 + k + offA + 8);
#pragma unroll
    for (int t = 0; t < 2; ++t) {
      v16bf br, bz, bx;
      ((v8bf*)&br)[0] = *(const v8bf*)(wgr[t] + GRU_I + k + offB);
      ((v8bf*)&br)[1] = *(const v8bf*)(wgr[t] + GRU_I + k + offB + 8);
      ((v8bf*)&bz)[0] = *(const v8bf*)(wgz[t] + GRU_I + k + offB);
      ((v8bf*)&bz)[1] = *(const v8bf*)(wgz[t] + GRU_I + k + offB + 8);
      ((v8bf*)&bx)[0] = *(const v8bf*)(whr[t] + k + offB);
      ((v8bf*)&bx)[1] = *(const v8bf*)(whr[t] + k + offB + 8);
      ar[0][t] = WMMA_BF16(a0, br, ar[0][t]);
      ar[1][t] = WMMA_BF16(a1, br, ar[1][t]);
      az[0][t] = WMMA_BF16(a0, bz, az[0][t]);
      az[1][t] = WMMA_BF16(a1, bz, az[1][t]);
      ah[0][t] = WMMA_BF16(a0, bx, ah[0][t]);
      ah[1][t] = WMMA_BF16(a1, bx, ah[1][t]);
    }
  }

  // ================= fused GRU epilogue =================
#pragma unroll
  for (int mh = 0; mh < 2; ++mh) {
#pragma unroll
    for (int t = 0; t < 2; ++t) {
      const int n = nidx[t];
      const float bgr = b_gate[n];
      const float bgz = b_gate[GRU_H + n];
      const float bii = b_i[n];
      const float bhh = b_h[n];
#pragma unroll
      for (int r = 0; r < 8; ++r) {
        const long  m  = m0 + mh * 16 + half * 8 + r;
        const float rg = fast_sigmoid(ar[mh][t][r] + bgr);
        const float zg = fast_sigmoid(az[mh][t][r] + bgz);
        const float ng = fast_tanh(ai[mh][t][r] + bii + rg * (ah[mh][t][r] + bhh));
        const float hp = hidden[m * GRU_H + n];
        out[m * GRU_H + n] = fmaf(zg, hp - ng, ng);   // (1-z)*n + z*h
      }
    }
  }
}

extern "C" void kernel_launch(void* const* d_in, const int* in_sizes, int n_in,
                              void* d_out, int out_size, void* d_ws, size_t ws_size,
                              hipStream_t stream) {
  const float* input  = (const float*)d_in[0];
  const float* hidden = (const float*)d_in[1];
  const float* W_gate = (const float*)d_in[2];
  const float* b_gate = (const float*)d_in[3];
  const float* W_i    = (const float*)d_in[4];
  const float* b_i    = (const float*)d_in[5];
  const float* W_h    = (const float*)d_in[6];
  const float* b_h    = (const float*)d_in[7];
  float* out = (float*)d_out;

  bf16_t* Wg = (bf16_t*)d_ws;                       // 1024*1024 bf16
  bf16_t* Wi = Wg + (long)GRU_K * (2 * GRU_H);      // 512*512 bf16
  bf16_t* Wh = Wi + (long)GRU_I * GRU_H;            // 512*512 bf16

  const long totalW = (long)GRU_K * (2 * GRU_H) + 2L * GRU_I * GRU_H;
  gru_cvt_weights<<<(unsigned)((totalW + 255) / 256), 256, 0, stream>>>(
      W_gate, W_i, W_h, Wg);

  gru_fused_wmma<<<dim3(GRU_B / 32, 2), 256, 0, stream>>>(
      input, hidden, Wg, Wi, Wh, b_gate, b_i, b_h, out);
}
